// GATLayer_2362232012921
// MI455X (gfx1250) — matrix operand order repaired and checked by reading, not compiled, not run on hardware
//
#include <hip/hip_runtime.h>
#include <hip/hip_bf16.h>
#include <math.h>

#define D_IN  128
#define D_OUT 64
#define NEG_SLOPE 0.2f
#define EPS_F 1e-10f

typedef __attribute__((ext_vector_type(16))) __bf16 v16bf;
typedef __attribute__((ext_vector_type(8)))  float  v8f;

// ---------------------------------------------------------------------------
// Pack W [128 x 64] fp32 (row-major, K x N) into bf16 WMMA B-fragment order:
// Wpk[((kt*4 + nt)*32 + lane)*16 + 2j+h] with
//   n = nt*16 + (lane&15), K = kt*32 + (lane>>4)*16 + 2j + h
// so the GEMM kernel reads each B fragment as one contiguous 32B v16bf load.
// ---------------------------------------------------------------------------
__global__ void gat_pack_W(const float* __restrict__ W, __bf16* __restrict__ Wpk) {
    int idx  = threadIdx.x;          // 0..511  (4 kt * 4 nt * 32 lanes)
    int kt   = idx >> 7;
    int nt   = (idx >> 5) & 3;
    int lane = idx & 31;
    int n     = nt * 16 + (lane & 15);
    int kbase = kt * 32 + (lane >> 4) * 16;
    __bf16* dst = Wpk + (size_t)idx * 16;
#pragma unroll
    for (int j = 0; j < 8; ++j) {
        int K0 = kbase + 2 * j;
        dst[2 * j]     = (__bf16)W[(size_t)K0 * D_OUT + n];
        dst[2 * j + 1] = (__bf16)W[(size_t)(K0 + 1) * D_OUT + n];
    }
}

// ---------------------------------------------------------------------------
// One wave computes a 16-row x 64-col tile of Wh = x @ W using
// v_wmma_f32_16x16x32_bf16 (4 N-tiles x 4 K-steps = 16 WMMA ops per wave).
// Epilogue fuses s_src = Wh . a_src and s_tgt = Wh . a_tgt via lane shuffles.
// ---------------------------------------------------------------------------
__global__ void gat_gemm_attn(const float* __restrict__ x,
                              const __bf16* __restrict__ Wpk,
                              const float* __restrict__ a,
                              float* __restrict__ Wh,
                              float* __restrict__ s_src,
                              float* __restrict__ s_tgt,
                              int N) {
    int wave   = blockIdx.x * (blockDim.x >> 5) + (threadIdx.x >> 5);
    int lane   = threadIdx.x & 31;
    int nTiles = (N + 15) >> 4;
    if (wave >= nTiles) return;            // wave-uniform: EXEC stays all-ones

    int rowBase = wave << 4;
    int m     = lane & 15;                 // column within 16-wide N tile / A row
    int khalf = lane >> 4;                 // which K half this lane holds for A
    int rowA  = rowBase + m;
    if (rowA >= N) rowA = N - 1;           // clamp for loads; stores are guarded

    v8f acc[4] = {v8f{}, v8f{}, v8f{}, v8f{}};

#pragma unroll
    for (int kt = 0; kt < 4; ++kt) {
        // A fragment: element s holds k = kt*32 + 16*khalf + s (contiguous, same order as Wpk)
        v16bf afrag;
        const float* xrow = x + (size_t)rowA * D_IN + kt * 32 + khalf * 16;
#pragma unroll
        for (int j = 0; j < 8; ++j) {
            int kOff = 2 * j;
            afrag[2 * j]     = (__bf16)xrow[kOff];
            afrag[2 * j + 1] = (__bf16)xrow[kOff + 1];
        }
#pragma unroll
        for (int nt = 0; nt < 4; ++nt) {
            v16bf bfrag = *(const v16bf*)(Wpk + (size_t)((kt * 4 + nt) * 32 + lane) * 16);
            acc[nt] = __builtin_amdgcn_wmma_f32_16x16x32_bf16(
                false, afrag, false, bfrag, (short)0, acc[nt], false, false);
        }
    }

    // attention-vector entries for this lane's columns
    float aS[4], aT[4];
#pragma unroll
    for (int nt = 0; nt < 4; ++nt) {
        int n  = nt * 16 + m;
        aS[nt] = a[n];
        aT[nt] = a[D_OUT + n];
    }

    // C/D layout: lanes 0-15 hold M=r, lanes 16-31 hold M=8+r, N=(lane&15)
#pragma unroll
    for (int r = 0; r < 8; ++r) {
        int rm = rowBase + khalf * 8 + r;
        float vs = 0.f, vt = 0.f;
#pragma unroll
        for (int nt = 0; nt < 4; ++nt) {
            float c = acc[nt][r];
            if (rm < N) Wh[(size_t)rm * D_OUT + nt * 16 + m] = c;
            vs += c * aS[nt];
            vt += c * aT[nt];
        }
#pragma unroll
        for (int off = 1; off < 16; off <<= 1) {     // reduce within each 16-lane half
            vs += __shfl_xor(vs, off, 32);
            vt += __shfl_xor(vt, off, 32);
        }
        if (m == 0 && rm < N) { s_src[rm] = vs; s_tgt[rm] = vt; }
    }
}

// ---------------------------------------------------------------------------
// init: out = 0, e_max = -inf, e_sum = 0
// ---------------------------------------------------------------------------
__global__ void gat_init(float* __restrict__ out, float* __restrict__ e_max,
                         float* __restrict__ e_sum, int N) {
    int i = blockIdx.x * blockDim.x + threadIdx.x;
    if (i < N * D_OUT) out[i] = 0.f;
    if (i < N) { e_max[i] = -INFINITY; e_sum[i] = 0.f; }
}

// Order-preserving float atomic max (no dependence on HIP float atomicMax):
// for v >= 0 the IEEE bit pattern is monotone as signed int (atomicMax),
// for v <  0 it is anti-monotone as unsigned int (atomicMin).
__device__ inline void atomicMaxF32(float* addr, float val) {
    if (val >= 0.f) atomicMax((int*)addr, __float_as_int(val));
    else            atomicMin((unsigned int*)addr, __float_as_uint(val));
}

// ---------------------------------------------------------------------------
// per-edge raw score + segment max over tgt
// ---------------------------------------------------------------------------
__global__ void gat_edge_score(const int* __restrict__ ei, const float* __restrict__ ew,
                               const float* __restrict__ s_src, const float* __restrict__ s_tgt,
                               float* __restrict__ e_buf, float* __restrict__ e_max, int E) {
    int j = blockIdx.x * blockDim.x + threadIdx.x;
    if (j >= E) return;
    int s = ei[j];
    int t = ei[E + j];
    float e = s_src[s] + s_tgt[t];
    e = (e > 0.f) ? e : NEG_SLOPE * e;
    e *= ew[j];
    e_buf[j] = e;
    atomicMaxF32(&e_max[t], e);
}

// ---------------------------------------------------------------------------
// e_exp = exp(e - e_max[tgt]); segment sum over tgt
// ---------------------------------------------------------------------------
__global__ void gat_edge_exp(const int* __restrict__ ei, float* __restrict__ e_buf,
                             const float* __restrict__ e_max, float* __restrict__ e_sum, int E) {
    int j = blockIdx.x * blockDim.x + threadIdx.x;
    if (j >= E) return;
    int t = ei[E + j];
    float ex = __expf(e_buf[j] - e_max[t]);
    e_buf[j] = ex;
    atomicAdd(&e_sum[t], ex);
}

// ---------------------------------------------------------------------------
// out[tgt] += (e_exp / (e_sum[tgt]+eps)) * Wh[src]
// 16 lanes per edge, 4 channels per lane (float4 gather, 4 atomic adds)
// ---------------------------------------------------------------------------
__global__ void gat_edge_scatter(const int* __restrict__ ei, const float* __restrict__ e_buf,
                                 const float* __restrict__ e_sum, const float* __restrict__ Wh,
                                 float* __restrict__ out, int E) {
    int gid = blockIdx.x * blockDim.x + threadIdx.x;
    int j   = gid >> 4;
    int c   = (gid & 15) * 4;
    if (j >= E) return;
    int s = ei[j];
    int t = ei[E + j];
    float alpha = e_buf[j] / (e_sum[t] + EPS_F);
    const float4 wh = *(const float4*)(Wh + (size_t)s * D_OUT + c);
    float* o = out + (size_t)t * D_OUT + c;
    atomicAdd(o + 0, alpha * wh.x);
    atomicAdd(o + 1, alpha * wh.y);
    atomicAdd(o + 2, alpha * wh.z);
    atomicAdd(o + 3, alpha * wh.w);
}

static inline size_t alignUp(size_t v, size_t a) { return (v + a - 1) & ~(a - 1); }

extern "C" void kernel_launch(void* const* d_in, const int* in_sizes, int n_in,
                              void* d_out, int out_size, void* d_ws, size_t ws_size,
                              hipStream_t stream) {
    (void)n_in; (void)out_size; (void)ws_size;
    const float* x  = (const float*)d_in[0];
    const int*   ei = (const int*)  d_in[1];   // [2, E]: row0 = src, row1 = tgt
    const float* ew = (const float*)d_in[2];
    const float* W  = (const float*)d_in[3];
    const float* a  = (const float*)d_in[4];
    float* out = (float*)d_out;

    const int N = in_sizes[0] / D_IN;
    const int E = in_sizes[2];

    // workspace carve (all 256B aligned)
    size_t off = 0;
    char* ws = (char*)d_ws;
    float* Wh    = (float*)(ws + off); off = alignUp(off + (size_t)N * D_OUT * sizeof(float), 256);
    float* s_src = (float*)(ws + off); off = alignUp(off + (size_t)N * sizeof(float), 256);
    float* s_tgt = (float*)(ws + off); off = alignUp(off + (size_t)N * sizeof(float), 256);
    float* e_max = (float*)(ws + off); off = alignUp(off + (size_t)N * sizeof(float), 256);
    float* e_sum = (float*)(ws + off); off = alignUp(off + (size_t)N * sizeof(float), 256);
    float* e_buf = (float*)(ws + off); off = alignUp(off + (size_t)E * sizeof(float), 256);
    __bf16* Wpk  = (__bf16*)(ws + off); // 4*4*32*16 bf16 = 16 KB

    // 1) pack W into WMMA B-fragment order
    gat_pack_W<<<1, 512, 0, stream>>>(W, Wpk);

    // 2) Wh = x @ W (bf16 WMMA, fp32 acc) + fused s_src/s_tgt
    int nTiles = (N + 15) >> 4;                 // 16 rows per wave
    int wavesPerBlock = 8;                      // 256 threads
    int gemmBlocks = (nTiles + wavesPerBlock - 1) / wavesPerBlock;
    gat_gemm_attn<<<gemmBlocks, 256, 0, stream>>>(x, Wpk, a, Wh, s_src, s_tgt, N);

    // 3) init out / e_max / e_sum
    int initThreads = N * D_OUT;
    gat_init<<<(initThreads + 255) / 256, 256, 0, stream>>>(out, e_max, e_sum, N);

    // 4) edge scores + segment max
    gat_edge_score<<<(E + 255) / 256, 256, 0, stream>>>(ei, ew, s_src, s_tgt, e_buf, e_max, E);

    // 5) exp + segment sum
    gat_edge_exp<<<(E + 255) / 256, 256, 0, stream>>>(ei, e_buf, e_max, e_sum, E);

    // 6) normalize + scatter-add messages
    long long scatterThreads = (long long)E * 16;
    int scatterBlocks = (int)((scatterThreads + 255) / 256);
    gat_edge_scatter<<<scatterBlocks, 256, 0, stream>>>(ei, e_buf, e_sum, Wh, out, E);
}
